// SplineCNN_5308579578323
// MI455X (gfx1250) — matrix operand order repaired and checked by reading, not compiled, not run on hardware
//
#include <hip/hip_runtime.h>
#include <hip/hip_bf16.h>
#include <math.h>

// ---------------------------------------------------------------------------
// SplineCNN forward for MI455X (gfx1250, wave32, WMMA).
//   L1: SplineConv(1->32)   -- scalar edge kernel + atomics (tiny FLOPs)
//   L2: SplineConv(32->64)  -- expanded-K GEMM via v_wmma_f32_16x16x32_f16
//   MLP: 64->128->10 + log_softmax -- per-node block kernel
// ---------------------------------------------------------------------------

typedef __attribute__((ext_vector_type(16))) _Float16 v16h;
typedef __attribute__((ext_vector_type(8)))  float    v8f;

#define KSZ 5
#define CIN 32
#define COUT 64

__device__ __forceinline__ float elu1(float v) {
    return v > 0.0f ? v : (expf(v) - 1.0f);
}

__device__ __forceinline__ void spline_pos(float ea, int& lo, float& fr) {
    float p = fminf(fmaxf(ea, 0.0f), 1.0f) * (float)(KSZ - 1);
    float pf = floorf(p);
    int l = (int)pf;
    if (l > KSZ - 2) l = KSZ - 2;
    if (l < 0) l = 0;
    lo = l;
    fr = p - (float)l;
}

// --------------------------------- zero ------------------------------------
__global__ void zero_f32(float* __restrict__ p, long n) {
    long stride = (long)gridDim.x * blockDim.x;
    for (long i = (long)blockIdx.x * blockDim.x + threadIdx.x; i < n; i += stride)
        p[i] = 0.0f;
}

// ------------------------- layer 1: edge scatter ---------------------------
// thread per (edge, out-channel); c==0 lane also accumulates degree.
__global__ void __launch_bounds__(256)
l1_edge(const float* __restrict__ x, const int* __restrict__ src,
        const int* __restrict__ dst, const float* __restrict__ ea,
        const float* __restrict__ W1, float* __restrict__ agg1,
        float* __restrict__ deg, int nE) {
    int tid = blockIdx.x * 256 + threadIdx.x;
    int e = tid >> 5;
    int c = tid & 31;
    if (e >= nE) return;
    int lo; float fr;
    spline_pos(ea[e], lo, fr);
    float xv = x[src[e]];
    // W1 shape (K,1,32)
    float w = (1.0f - fr) * W1[lo * 32 + c] + fr * W1[(lo + 1) * 32 + c];
    int d = dst[e];
    atomicAdd(&agg1[d * 32 + c], xv * w);
    if (c == 0) atomicAdd(&deg[d], 1.0f);
}

// ------------------------- layer 1: node update ----------------------------
__global__ void __launch_bounds__(256)
l1_node(const float* __restrict__ x, const float* __restrict__ root1,
        const float* __restrict__ b1, const float* __restrict__ agg1,
        const float* __restrict__ deg, float* __restrict__ h1, int nN) {
    int tid = blockIdx.x * 256 + threadIdx.x;
    int n = tid >> 5;
    int c = tid & 31;
    if (n >= nN) return;
    float a = agg1[n * 32 + c] / fmaxf(deg[n], 1.0f);
    h1[n * 32 + c] = elu1(a + x[n] * root1[c] + b1[c]);
}

// --------------------- layer 2: edge GEMM via WMMA -------------------------
// Expanded-K formulation: A[e][k*32+c] = coef_k(e) * h1[src[e]][c]  (16x160)
// B = W2 flattened (160x64).  Per 16-edge tile: 5 k-steps x 4 n-tiles of
// v_wmma_f32_16x16x32_f16, f32 accumulate, then atomic scatter on dst.
__global__ void __launch_bounds__(256)
l2_edge_wmma(const float* __restrict__ h1, const int* __restrict__ src,
             const int* __restrict__ dst, const float* __restrict__ ea,
             const float* __restrict__ W2, float* __restrict__ agg2, int nE) {
    // B fragments, fragment-major: [ks][t][lane][16 halfs] = 20 KB
    __shared__ __align__(32) _Float16 sWf[KSZ * 4 * 32 * 16];
    // A tiles, per wave: [m][ks][32 perm-halfs] = 8 * 5 KB = 40 KB
    __shared__ __align__(32) _Float16 sA[8][16 * KSZ * 32];
    __shared__ int sDst[8][16];

    const int tid = threadIdx.x;
    const int wave = tid >> 5;
    const int lane = tid & 31;

    // ---- stage W2 (f32 global) -> f16 B fragments in LDS ----
    // B layout used here (contiguous K, not the ISA table order): lanes 0-15 hold col n=lane, K=0..15;
    // lanes 16-31 hold col n=lane-16, K=16..31 (elements sequential in K).
    for (int idx = tid; idx < KSZ * 4 * 32 * 16; idx += 256) {
        int i  = idx & 15;
        int l  = (idx >> 4) & 31;
        int t  = (idx >> 9) & 3;
        int ks = idx >> 11;
        int n   = (l & 15) + t * 16;
        int cin = ((l < 16) ? 0 : 16) + i;
        sWf[idx] = (_Float16)W2[(ks * 32 + cin) * 64 + n];   // W2 (K,32,64)
    }

    // ---- pack A tile (16 edges x 160) into per-wave LDS ----
    const int tile  = blockIdx.x * 8 + wave;
    const int ebase = tile * 16;
    const int m     = lane & 15;          // edge row within tile
    const int e     = ebase + m;
    const bool valid = (e < nE);

    int s = 0, d = 0, lo = 0;
    float fr = 0.0f;
    if (valid) {
        s = src[e];
        d = dst[e];
        spline_pos(ea[e], lo, fr);
    }
    if (lane < 16) sDst[wave][m] = d;

    // lanes 0-15 pack channels 0..15 of edge m; lanes 16-31 pack 16..31.
    const int c0 = (lane < 16) ? 0 : 16;
    float xs[16];
#pragma unroll
    for (int j = 0; j < 16; ++j) xs[j] = valid ? h1[s * 32 + c0 + j] : 0.0f;

    _Float16* aRow = &sA[wave][m * (KSZ * 32)];
#pragma unroll
    for (int ks = 0; ks < KSZ; ++ks) {
        float coef = 0.0f;
        if (valid) {
            if (ks == lo)          coef = 1.0f - fr;
            else if (ks == lo + 1) coef = fr;
        }
#pragma unroll
        for (int j = 0; j < 16; ++j) {
            int c = c0 + j;
            // A-layout within a 32-wide k-step (contiguous, matches B):
            // positions 0..15 hold K 0-15 (lanes 0-15),
            // positions 16..31 hold K 16-31 (lanes 16-31).
            int p = c;
            aRow[ks * 32 + p] = (_Float16)(coef * xs[j]);
        }
    }
    __syncthreads();

    // ---- 20 WMMAs: 5 k-steps x 4 n-tiles, f32 accumulate ----
    v8f acc[4] = {};
    const int aoff = (lane < 16) ? 0 : 16;
#pragma unroll
    for (int ks = 0; ks < KSZ; ++ks) {
        v16h a = *(const v16h*)&sA[wave][(m * KSZ + ks) * 32 + aoff];
#pragma unroll
        for (int t = 0; t < 4; ++t) {
            v16h b = *(const v16h*)&sWf[((ks * 4 + t) * 32 + lane) * 16];
            acc[t] = __builtin_amdgcn_wmma_f32_16x16x32_f16(
                false, a, false, b, (short)0, acc[t], false, false);
        }
    }

    // ---- scatter: C/D layout: lane -> n = lane%16, VGPR j -> m = j + 8*(lane/16)
    const int n     = lane & 15;
    const int mbase = (lane < 16) ? 0 : 8;
#pragma unroll
    for (int t = 0; t < 4; ++t) {
#pragma unroll
        for (int j = 0; j < 8; ++j) {
            int mm = mbase + j;
            if (ebase + mm < nE)
                atomicAdd(&agg2[sDst[wave][mm] * 64 + t * 16 + n], acc[t][j]);
        }
    }
}

// ------------------------- layer 2: node update ----------------------------
__global__ void __launch_bounds__(256)
l2_node(const float* __restrict__ h1, const float* __restrict__ root2,
        const float* __restrict__ b2, const float* __restrict__ agg2,
        const float* __restrict__ deg, float* __restrict__ h2, int nN) {
    int tid = blockIdx.x * 256 + threadIdx.x;
    int n = tid >> 6;
    int o = tid & 63;
    if (n >= nN) return;
    float acc = agg2[n * 64 + o] / fmaxf(deg[n], 1.0f) + b2[o];
#pragma unroll 8
    for (int i = 0; i < 32; ++i) acc += h1[n * 32 + i] * root2[i * 64 + o];
    h2[n * 64 + o] = elu1(acc);
}

// ---------------------- MLP + log_softmax (per node) -----------------------
__global__ void __launch_bounds__(128)
mlp_head(const float* __restrict__ h2, const float* __restrict__ lw1,
         const float* __restrict__ lb1, const float* __restrict__ lw2,
         const float* __restrict__ lb2, float* __restrict__ out, int nN) {
    __shared__ float sh[64];
    __shared__ float sf[128];
    __shared__ float so[10];
    __shared__ float red[2];
    int n = blockIdx.x;
    int t = threadIdx.x;
    if (n >= nN) return;
    if (t < 64) sh[t] = h2[n * 64 + t];
    __syncthreads();
    float f = lb1[t];
#pragma unroll 8
    for (int i = 0; i < 64; ++i) f += sh[i] * lw1[i * 128 + t];
    sf[t] = elu1(f);
    __syncthreads();
    if (t < 10) {
        float o = lb2[t];
#pragma unroll 8
        for (int i = 0; i < 128; ++i) o += sf[i] * lw2[i * 10 + t];
        so[t] = o;
    }
    __syncthreads();
    if (t == 0) {
        float mx = so[0];
        for (int i = 1; i < 10; ++i) mx = fmaxf(mx, so[i]);
        float sum = 0.0f;
        for (int i = 0; i < 10; ++i) sum += expf(so[i] - mx);
        red[0] = mx;
        red[1] = logf(sum);
    }
    __syncthreads();
    if (t < 10) out[n * 10 + t] = so[t] - red[0] - red[1];
}

// ---------------------------------------------------------------------------
extern "C" void kernel_launch(void* const* d_in, const int* in_sizes, int n_in,
                              void* d_out, int out_size, void* d_ws, size_t ws_size,
                              hipStream_t stream) {
    const float* x     = (const float*)d_in[0];
    const int*   eidx  = (const int*)d_in[1];
    const float* ea    = (const float*)d_in[2];
    const float* W1    = (const float*)d_in[3];
    const float* root1 = (const float*)d_in[4];
    const float* b1    = (const float*)d_in[5];
    const float* W2    = (const float*)d_in[6];
    const float* root2 = (const float*)d_in[7];
    const float* b2    = (const float*)d_in[8];
    const float* lw1   = (const float*)d_in[9];
    const float* lb1   = (const float*)d_in[10];
    const float* lw2   = (const float*)d_in[11];
    const float* lb2   = (const float*)d_in[12];
    float* out = (float*)d_out;

    const int nN = in_sizes[0];        // x is (N,1)
    const int nE = in_sizes[2];        // edge_attr is (E,1)
    const int* src = eidx;
    const int* dst = eidx + nE;

    // workspace layout (floats): deg | agg1 | agg2 | h1 | h2  (~38.6 MB)
    float* deg  = (float*)d_ws;
    float* agg1 = deg  + (size_t)nN;
    float* agg2 = agg1 + (size_t)nN * 32;
    float* h1   = agg2 + (size_t)nN * 64;
    float* h2   = h1   + (size_t)nN * 32;

    // zero accumulators (deg|agg1|agg2 are contiguous: N*97 floats)
    zero_f32<<<2048, 256, 0, stream>>>(deg, (long)nN * 97);

    // layer 1
    {
        long threads = (long)nE * 32;
        int blocks = (int)((threads + 255) / 256);
        l1_edge<<<blocks, 256, 0, stream>>>(x, src, dst, ea, W1, agg1, deg, nE);
    }
    {
        long threads = (long)nN * 32;
        int blocks = (int)((threads + 255) / 256);
        l1_node<<<blocks, 256, 0, stream>>>(x, root1, b1, agg1, deg, h1, nN);
    }

    // layer 2 (WMMA edge GEMM)
    {
        int tiles  = (nE + 15) / 16;
        int blocks = (tiles + 7) / 8;
        l2_edge_wmma<<<blocks, 256, 0, stream>>>(h1, src, dst, ea, W2, agg2, nE);
    }
    {
        long threads = (long)nN * 64;
        int blocks = (int)((threads + 255) / 256);
        l2_node<<<blocks, 256, 0, stream>>>(h1, root2, b2, agg2, deg, h2, nN);
    }

    // MLP head + log_softmax
    mlp_head<<<nN, 128, 0, stream>>>(h2, lw1, lb1, lw2, lb2, out, nN);
}
